// VQVAE_20693152432859
// MI455X (gfx1250) — matrix operand order repaired and checked by reading, not compiled, not run on hardware
//
#include <hip/hip_runtime.h>
#include <hip/hip_bf16.h>
#include <stdint.h>

typedef __attribute__((ext_vector_type(16))) _Float16 v16h;
typedef __attribute__((ext_vector_type(8)))  float    v8f;
typedef __attribute__((ext_vector_type(4)))  float    v4f;

#define BATCH   32768
#define IN_DIM  1024
#define HID     128
#define LATD    64
#define NEMB    4096

// -------- workspace layout (dword offsets) --------
#define W1F_OFF 0         // enc_w1 frags: 32 ksteps * 8 ntiles * 256 = 65536 dwords
#define W2F_OFF 65536     // enc_w2 frags:  4 * 4 * 256 = 4096
#define W3F_OFF 69632     // dec_w1 frags:  2 * 8 * 256 = 4096
#define W4F_OFF 73728     // dec_w2 frags:  4 * 64 * 256 = 65536
#define CBF_OFF 139264    // codebook^T frags (chunk-major): 256 tiles * 2 ksteps * 256 = 131072
#define E2_OFF  270336    // ||e||^2 per entry: 4096 floats

// -------- output layout (float offsets) --------
#define O_LAT  ((size_t)0)
#define O_Q    ((size_t)BATCH * LATD)
#define O_REC  ((size_t)2 * BATCH * LATD)
#define O_LOSS ((size_t)2 * BATCH * LATD + (size_t)BATCH * IN_DIM)

// ---------------- helpers ----------------

union F16x16 { v16h v; uint4 q[2]; };

__device__ __forceinline__ uint32_t pack2(float a, float b) {
  union { _Float16 h[2]; uint32_t u; } r;
  r.h[0] = (_Float16)a;
  r.h[1] = (_Float16)b;
  return r.u;
}

// B fragment (32x16 K x N, f16) pre-packed: [(kstep*NT + tile)*32 + lane]*8 dwords
__device__ __forceinline__ v16h load_b_frag(const uint32_t* __restrict__ frag,
                                            int kstep, int NT, int tile, int lane) {
  F16x16 r;
  const uint32_t* p = frag + ((size_t)((kstep * NT + tile) * 32 + lane) << 3);
  r.q[0] = *(const uint4*)p;
  r.q[1] = *(const uint4*)(p + 4);
  return r.v;
}

// B fragment from an LDS-staged dword array (8 consecutive dwords per lane)
__device__ __forceinline__ v16h load_b_lds(const uint32_t* p) {
  F16x16 r;
  r.q[0] = *(const uint4*)p;
  r.q[1] = *(const uint4*)(p + 4);
  return r.v;
}

// A fragment (16x32 M x K, f16) from row-major f16 LDS tile [16][K].
__device__ __forceinline__ v16h load_a_lds(const _Float16* base, int row, int K,
                                           int kstep, int g) {
  F16x16 r;
  const _Float16* p = base + row * K + kstep * 32 + g * 8;
  r.q[0] = *(const uint4*)p;
  r.q[1] = *(const uint4*)(p + 16);
  return r.v;
}

// A fragment built from a global row-major f32 row (streaming: non-temporal loads)
__device__ __forceinline__ v16h load_a_x(const float* __restrict__ rowp, int kstep, int g) {
  const v4f* p0 = (const v4f*)(rowp + kstep * 32 + g * 8);
  const v4f* p1 = (const v4f*)(rowp + kstep * 32 + 16 + g * 8);
  v4f a = __builtin_nontemporal_load(p0);
  v4f b = __builtin_nontemporal_load(p0 + 1);
  v4f c = __builtin_nontemporal_load(p1);
  v4f d = __builtin_nontemporal_load(p1 + 1);
  F16x16 r;
#pragma unroll
  for (int i = 0; i < 4; ++i) {
    r.v[i]      = (_Float16)a[i];
    r.v[4 + i]  = (_Float16)b[i];
    r.v[8 + i]  = (_Float16)c[i];
    r.v[12 + i] = (_Float16)d[i];
  }
  return r.v;
}

__device__ __forceinline__ v8f wmma16(v16h a, v16h b, v8f c) {
  return __builtin_amdgcn_wmma_f32_16x16x32_f16(false, a, false, b, (short)0, c,
                                                false, false);
}

// ---------------- prep kernels ----------------

// Pack K x N f32 (element (k,n) at src[k*sk + n*sn]) into WMMA B-fragment layout, f16.
__global__ void prep_frag(const float* __restrict__ src, uint32_t* __restrict__ dst,
                          int NT, int sk, int sn, int total) {
  int id = blockIdx.x * blockDim.x + threadIdx.x;
  if (id >= total) return;
  int v = id & 7;
  int lane = (id >> 3) & 31;
  int tl = id >> 8;
  int t = tl % NT;
  int s = tl / NT;
  int n = t * 16 + (lane & 15);
  int g = lane >> 4;
  int k0 = s * 32 + (v >> 2) * 16 + g * 8 + 2 * (v & 3);
  dst[id] = pack2(src[(size_t)k0 * sk + (size_t)n * sn],
                  src[(size_t)(k0 + 1) * sk + (size_t)n * sn]);
}

// Codebook^T fragments in CHUNK-MAJOR order so each 16-entry chunk's two ksteps are
// a contiguous 2 KB span (async-copy friendly): index ((tile*2 + kstep)*32+lane)*8+v.
__global__ void prep_cb_frag(const float* __restrict__ cb, uint32_t* __restrict__ dst) {
  int id = blockIdx.x * blockDim.x + threadIdx.x;  // total 131072
  int v = id & 7;
  int lane = (id >> 3) & 31;
  int tl = id >> 8;        // tile*2 + kstep
  int tile = tl >> 1;
  int s = tl & 1;
  int n = tile * 16 + (lane & 15);   // codebook entry
  int g = lane >> 4;
  int k0 = s * 32 + (v >> 2) * 16 + g * 8 + 2 * (v & 3);  // latent dim
  dst[id] = pack2(cb[(size_t)n * LATD + k0], cb[(size_t)n * LATD + k0 + 1]);
}

// codebook row norms + zero the loss accumulator (re-done every launch: graph-safe)
__global__ void prep_e2(const float* __restrict__ cb, float* __restrict__ e2,
                        float* __restrict__ lossp) {
  int id = blockIdx.x * blockDim.x + threadIdx.x;
  if (id == 0) *lossp = 0.0f;
  if (id < NEMB) {
    float s = 0.f;
#pragma unroll 8
    for (int d = 0; d < LATD; ++d) {
      float v = cb[(size_t)id * LATD + d];
      s += v * v;
    }
    e2[id] = s;
  }
}

// ---------------- fused main kernel ----------------

struct __align__(16) WaveSmem {
  _Float16 hb[16 * HID];   // 4096 B: h / h2 tile (reused)
  _Float16 lat[16 * LATD]; // 2048 B: latent f16 tile
  _Float16 qb[16 * LATD];  // 2048 B: quantized f16 tile
  int idx[16];             // argmin indices
};

#define VQ_STAGE_CHUNKS 8                 // 8 x 16 entries per stage
#define VQ_STAGE_DW     (VQ_STAGE_CHUNKS * 2 * 256)   // 4096 dwords = 16 KB
#define VQ_NSTAGES      (NEMB / 16 / VQ_STAGE_CHUNKS) // 32

__global__ __launch_bounds__(256) void vqvae_main(
    const float* __restrict__ x, const float* __restrict__ eb1,
    const float* __restrict__ eb2, const float* __restrict__ cb,
    const float* __restrict__ db1, const float* __restrict__ db2,
    const uint32_t* __restrict__ ws, float* __restrict__ out) {
  __shared__ WaveSmem sm[8];
  __shared__ __align__(16) uint32_t cb_lds[2][VQ_STAGE_DW];  // 32 KB double buffer

  const int lane = threadIdx.x & 31;
  const int wave = threadIdx.x >> 5;
  const int lid  = lane & 15;   // C-frag column / A-frag row
  const int g    = lane >> 4;   // half-wave group
  const int R    = (blockIdx.x * 8 + wave) * 16;  // 16-row tile base
  WaveSmem& S = sm[wave];

  const uint32_t* W1f = ws + W1F_OFF;
  const uint32_t* W2f = ws + W2F_OFF;
  const uint32_t* W3f = ws + W3F_OFF;
  const uint32_t* W4f = ws + W4F_OFF;
  const float*    e2  = (const float*)(ws + E2_OFF);

  // ========== encoder GEMM1: h[16x128] = relu(x[16x1024] @ W1 + b1) ==========
  v8f acc[8];
#pragma unroll
  for (int t = 0; t < 8; ++t)
#pragma unroll
    for (int j = 0; j < 8; ++j) acc[t][j] = 0.f;

  const float* xrow = x + (size_t)(R + lid) * IN_DIM;
  for (int s = 0; s < IN_DIM / 32; ++s) {
    v16h a = load_a_x(xrow, s, g);
#pragma unroll
    for (int t = 0; t < 8; ++t)
      acc[t] = wmma16(a, load_b_frag(W1f, s, 8, t, lane), acc[t]);
  }
#pragma unroll
  for (int t = 0; t < 8; ++t) {
    float bias = eb1[t * 16 + lid];
#pragma unroll
    for (int v = 0; v < 8; ++v) {
      float val = fmaxf(acc[t][v] + bias, 0.f);
      S.hb[(v + 8 * g) * HID + t * 16 + lid] = (_Float16)val;
    }
  }
  __syncthreads();

  // ========== encoder GEMM2: latent[16x64] = relu(h @ W2 + b2) ==========
  v8f acc2[4];
#pragma unroll
  for (int t = 0; t < 4; ++t)
#pragma unroll
    for (int j = 0; j < 8; ++j) acc2[t][j] = 0.f;
#pragma unroll
  for (int s = 0; s < 4; ++s) {
    v16h a = load_a_lds(S.hb, lid, HID, s, g);
#pragma unroll
    for (int t = 0; t < 4; ++t)
      acc2[t] = wmma16(a, load_b_frag(W2f, s, 4, t, lane), acc2[t]);
  }
#pragma unroll
  for (int t = 0; t < 4; ++t) {
    float bias = eb2[t * 16 + lid];
#pragma unroll
    for (int v = 0; v < 8; ++v) {
      float val = fmaxf(acc2[t][v] + bias, 0.f);
      __builtin_nontemporal_store(
          val, out + O_LAT + (size_t)(R + v + 8 * g) * LATD + t * 16 + lid);
      S.lat[(v + 8 * g) * LATD + t * 16 + lid] = (_Float16)val;
    }
  }
  __syncthreads();

  // ========== VQ: argmin_k ||latent - e_k||^2 == argmin (e2 - 2*dot) ==========
  // Codebook fragments are staged block-cooperatively through LDS using the CDNA5
  // async global->LDS path (one copy per block instead of one per wave), double
  // buffered and overlapped with the WMMA chunk loop.
  v16h a0 = load_a_lds(S.lat, lid, LATD, 0, g);
  v16h a1 = load_a_lds(S.lat, lid, LATD, 1, g);
  float mval[8];
  int midx[8];
#pragma unroll
  for (int v = 0; v < 8; ++v) { mval[v] = 3.0e38f; midx[v] = 0; }

  const uint64_t cbf_g = (uint64_t)(uintptr_t)(ws + CBF_OFF);
  const uint32_t cb_l  = (uint32_t)(uintptr_t)(&cb_lds[0][0]);
  // each async b128 op moves 32 lanes * 16 B = 512 B contiguous
  const uint32_t seg = (uint32_t)(wave * 2048 + lane * 16);

  auto issue_stage = [&](int st, int buf) {
    uint64_t gaddr = cbf_g + (uint64_t)st * (VQ_STAGE_DW * 4) + seg;
    uint32_t laddr = cb_l + (uint32_t)buf * (VQ_STAGE_DW * 4) + seg;
#pragma unroll
    for (int i = 0; i < 4; ++i) {
      asm volatile("global_load_async_to_lds_b128 %0, %1, off"
                   :: "v"(laddr + i * 512), "v"(gaddr + (uint64_t)(i * 512))
                   : "memory");
    }
  };

  issue_stage(0, 0);
  for (int st = 0; st < VQ_NSTAGES; ++st) {
    asm volatile("s_wait_asynccnt 0" ::: "memory");
    __syncthreads();
    if (st + 1 < VQ_NSTAGES) issue_stage(st + 1, (st + 1) & 1);
    const uint32_t* bb = cb_lds[st & 1];
#pragma unroll
    for (int cc = 0; cc < VQ_STAGE_CHUNKS; ++cc) {
      int c = st * VQ_STAGE_CHUNKS + cc;
      v8f dot;
#pragma unroll
      for (int j = 0; j < 8; ++j) dot[j] = 0.f;
      dot = wmma16(a0, load_b_lds(bb + ((cc * 2 + 0) * 32 + lane) * 8), dot);
      dot = wmma16(a1, load_b_lds(bb + ((cc * 2 + 1) * 32 + lane) * 8), dot);
      int entry = c * 16 + lid;
      float e2v = e2[entry];
#pragma unroll
      for (int v = 0; v < 8; ++v) {
        float d = e2v - 2.f * dot[v];
        if (d < mval[v]) { mval[v] = d; midx[v] = entry; }
      }
    }
  }
  // reduce across the 16 lanes holding the same row (masks stay within half-wave)
#pragma unroll
  for (int off = 8; off >= 1; off >>= 1) {
#pragma unroll
    for (int v = 0; v < 8; ++v) {
      float ov = __shfl_xor(mval[v], off, 32);
      int oi = __shfl_xor(midx[v], off, 32);
      if (ov < mval[v] || (ov == mval[v] && oi < midx[v])) {
        mval[v] = ov;
        midx[v] = oi;
      }
    }
  }
  if (lid == 0) {
#pragma unroll
    for (int v = 0; v < 8; ++v) S.idx[v + 8 * g] = midx[v];
  }
  __syncthreads();

  // ========== gather quantized rows, emit quantized_st, accumulate loss ==========
  float lsum = 0.f;
#pragma unroll 4
  for (int r = 0; r < 16; ++r) {
    int e = S.idx[r];
    float qa = cb[(size_t)e * LATD + lane];
    float qb = cb[(size_t)e * LATD + 32 + lane];
    size_t orow = O_Q + (size_t)(R + r) * LATD;
    __builtin_nontemporal_store(qa, out + orow + lane);
    __builtin_nontemporal_store(qb, out + orow + 32 + lane);
    S.qb[r * LATD + lane] = (_Float16)qa;
    S.qb[r * LATD + 32 + lane] = (_Float16)qb;
    float la = (float)S.lat[r * LATD + lane];
    float lb = (float)S.lat[r * LATD + 32 + lane];
    lsum += (qa - la) * (qa - la) + (qb - lb) * (qb - lb);
  }
#pragma unroll
  for (int off = 16; off >= 1; off >>= 1) lsum += __shfl_xor(lsum, off, 32);
  if (lane == 0)
    atomicAdd(out + O_LOSS, lsum * (1.25f / ((float)BATCH * (float)LATD)));
  __syncthreads();

  // ========== decoder GEMM3: h2[16x128] = relu(q @ W3 + db1) ==========
#pragma unroll
  for (int t = 0; t < 8; ++t)
#pragma unroll
    for (int j = 0; j < 8; ++j) acc[t][j] = 0.f;
#pragma unroll
  for (int s = 0; s < 2; ++s) {
    v16h a = load_a_lds(S.qb, lid, LATD, s, g);
#pragma unroll
    for (int t = 0; t < 8; ++t)
      acc[t] = wmma16(a, load_b_frag(W3f, s, 8, t, lane), acc[t]);
  }
#pragma unroll
  for (int t = 0; t < 8; ++t) {
    float bias = db1[t * 16 + lid];
#pragma unroll
    for (int v = 0; v < 8; ++v) {
      float val = fmaxf(acc[t][v] + bias, 0.f);
      S.hb[(v + 8 * g) * HID + t * 16 + lid] = (_Float16)val;
    }
  }
  __syncthreads();

  // ========== decoder GEMM4: recon[16x1024] = sigmoid(h2 @ W4 + db2) ==========
  v16h a4[4];
#pragma unroll
  for (int s = 0; s < 4; ++s) a4[s] = load_a_lds(S.hb, lid, HID, s, g);

  for (int t = 0; t < IN_DIM / 16; ++t) {
    float bias = db2[t * 16 + lid];
    v8f c;
#pragma unroll
    for (int j = 0; j < 8; ++j) c[j] = bias;   // init accumulator with bias
#pragma unroll
    for (int s = 0; s < 4; ++s)
      c = wmma16(a4[s], load_b_frag(W4f, s, IN_DIM / 16, t, lane), c);
#pragma unroll
    for (int v = 0; v < 8; ++v) {
      float val = 1.f / (1.f + __expf(-c[v]));
      __builtin_nontemporal_store(
          val, out + O_REC + (size_t)(R + v + 8 * g) * IN_DIM + t * 16 + lid);
    }
  }
}

// ---------------- launcher ----------------

extern "C" void kernel_launch(void* const* d_in, const int* in_sizes, int n_in,
                              void* d_out, int out_size, void* d_ws, size_t ws_size,
                              hipStream_t stream) {
  const float* x   = (const float*)d_in[0];
  const float* ew1 = (const float*)d_in[1];
  const float* eb1 = (const float*)d_in[2];
  const float* ew2 = (const float*)d_in[3];
  const float* eb2 = (const float*)d_in[4];
  const float* cb  = (const float*)d_in[5];
  const float* dw1 = (const float*)d_in[6];
  const float* db1 = (const float*)d_in[7];
  const float* dw2 = (const float*)d_in[8];
  const float* db2 = (const float*)d_in[9];
  uint32_t* ws = (uint32_t*)d_ws;
  float* out = (float*)d_out;

  // Pack weights + codebook^T into WMMA B-fragment layout (f16), per launch.
  prep_frag<<<65536 / 256, 256, 0, stream>>>(ew1, ws + W1F_OFF, 8, HID, 1, 65536);
  prep_frag<<<4096 / 256, 256, 0, stream>>>(ew2, ws + W2F_OFF, 4, LATD, 1, 4096);
  prep_frag<<<4096 / 256, 256, 0, stream>>>(dw1, ws + W3F_OFF, 8, HID, 1, 4096);
  prep_frag<<<65536 / 256, 256, 0, stream>>>(dw2, ws + W4F_OFF, 64, IN_DIM, 1, 65536);
  prep_cb_frag<<<131072 / 256, 256, 0, stream>>>(cb, ws + CBF_OFF);
  prep_e2<<<NEMB / 256, 256, 0, stream>>>(cb, (float*)(ws + E2_OFF), out + O_LOSS);

  // Fused encoder -> VQ -> decoder: 8 waves/block, 16 rows/wave, 128 rows/block.
  vqvae_main<<<BATCH / 128, 256, 0, stream>>>(x, eb1, eb2, cb, db1, db2, ws, out);
}
